// LSTM_20744692039700
// MI455X (gfx1250) — matrix operand order repaired and checked by reading, not compiled, not run on hardware
//
#include <hip/hip_runtime.h>
#include <hip/hip_bf16.h>
#include <stdint.h>

// ---------------------------------------------------------------------------
// LSTM LM: embed -> 256-step LSTM (fused GEMM+gates per step) -> tied proj.
// bf16 WMMA (v_wmma_f32_16x16x32_bf16) with f32 accumulation throughout.
// Projection stages B tiles into LDS via GLOBAL_LOAD_ASYNC_TO_LDS_B128
// (ASYNCcnt-tracked), double buffered, 8 N-tiles per wave.
// ---------------------------------------------------------------------------

typedef __attribute__((ext_vector_type(16))) __bf16 v16bf;
typedef __attribute__((ext_vector_type(8)))  float  v8f;
typedef int v4i_ __attribute__((vector_size(16)));   // matches builtin proto

#define B_ 16
#define S_ 256
#define V_ 32000
#define H_ 1024

#ifndef __has_builtin
#define __has_builtin(x) 0
#endif

#if __has_builtin(__builtin_amdgcn_global_load_async_to_lds_b128)
#define USE_ASYNC_LDS 1
#else
#define USE_ASYNC_LDS 0
#endif

#if USE_ASYNC_LDS
#if __has_builtin(__builtin_amdgcn_s_wait_asynccnt)
#define ASYNC_WAIT() __builtin_amdgcn_s_wait_asynccnt(0)
#else
#define ASYNC_WAIT() asm volatile("s_wait_asynccnt 0" ::: "memory")
#endif
#else
#define ASYNC_WAIT()
#endif

// 16-byte global -> LDS copy (async on gfx1250).
__device__ __forceinline__ void copy16_g2l(const __bf16* g, __bf16* l) {
#if USE_ASYNC_LDS
    __builtin_amdgcn_global_load_async_to_lds_b128(
        (__attribute__((address_space(1))) v4i_*)(uintptr_t)g,
        (__attribute__((address_space(3))) v4i_*)(uint32_t)(uintptr_t)l,
        0, 0);
#else
    *(uint4*)l = *(const uint4*)g;
#endif
}

// Workspace layout (bytes). Total ~99.3 MB.
static constexpr size_t OFF_EMBBF = 0;                           // [V,H] bf16
static constexpr size_t SZ_EMBBF  = (size_t)V_ * H_ * 2;
static constexpr size_t OFF_WBF   = OFF_EMBBF + SZ_EMBBF;        // [4H,2H] bf16
static constexpr size_t SZ_WBF    = (size_t)4 * H_ * 2 * H_ * 2;
static constexpr size_t OFF_XEBF  = OFF_WBF + SZ_WBF;            // [S,B,H] bf16
static constexpr size_t SZ_XEBF   = (size_t)S_ * B_ * H_ * 2;
static constexpr size_t OFF_YSBF  = OFF_XEBF + SZ_XEBF;          // [S,B,H] bf16
static constexpr size_t SZ_YSBF   = SZ_XEBF;
static constexpr size_t OFF_HBF   = OFF_YSBF + SZ_YSBF;          // 2x[B,H] bf16
static constexpr size_t SZ_HBF    = (size_t)2 * B_ * H_ * 2;
static constexpr size_t OFF_CST   = OFF_HBF + SZ_HBF;            // [B,H] f32

struct bf16x4 { __bf16 x, y, z, w; };

__device__ __forceinline__ bf16x4 cvt4(float4 f) {
    bf16x4 r; r.x = (__bf16)f.x; r.y = (__bf16)f.y; r.z = (__bf16)f.z; r.w = (__bf16)f.w;
    return r;
}
__device__ __forceinline__ float sigmoidf_(float x) {
    return 1.0f / (1.0f + __expf(-x));
}

// ---- one-time prep kernels -------------------------------------------------

__global__ void cast_bf16_kernel(const float* __restrict__ src,
                                 __bf16* __restrict__ dst, int n4) {
    int i = blockIdx.x * blockDim.x + threadIdx.x;
    if (i < n4) {
        float4 f = ((const float4*)src)[i];
        ((bf16x4*)dst)[i] = cvt4(f);
    }
}

// Pack W_i/W_f/W_o/W_c ([H,2H] f32 each) into one [4H,2H] bf16 matrix.
__global__ __launch_bounds__(256) void pack_w_kernel(
        const float* __restrict__ Wi, const float* __restrict__ Wf,
        const float* __restrict__ Wo, const float* __restrict__ Wc,
        __bf16* __restrict__ dst) {
    int n = blockIdx.x;                 // 0..4095
    int g = n >> 10;
    int row = n & (H_ - 1);
    const float* src = (g == 0) ? Wi : (g == 1) ? Wf : (g == 2) ? Wo : Wc;
    src += (size_t)row * (2 * H_);
    __bf16* d = dst + (size_t)n * (2 * H_);
    int c0 = threadIdx.x * 8;
    float4 f0 = *(const float4*)(src + c0);
    float4 f1 = *(const float4*)(src + c0 + 4);
    *(bf16x4*)(d + c0)     = cvt4(f0);
    *(bf16x4*)(d + c0 + 4) = cvt4(f1);
}

// Gather embedding rows into time-major bf16 activations xe[s][b][h].
__global__ __launch_bounds__(256) void embed_kernel(
        const int* __restrict__ x, const float* __restrict__ emb,
        __bf16* __restrict__ xe) {
    int bid = blockIdx.x;               // s*16 + b
    int s = bid >> 4, b = bid & 15;
    int tok = x[b * S_ + s];
    int h0 = threadIdx.x * 4;
    float4 f = *(const float4*)(emb + (size_t)tok * H_ + h0);
    *(bf16x4*)(xe + (size_t)bid * H_ + h0) = cvt4(f);
}

__global__ void init_state_kernel(float* __restrict__ c, __bf16* __restrict__ h0) {
    int i = blockIdx.x * blockDim.x + threadIdx.x;
    if (i < B_ * H_) { c[i] = 0.0f; h0[i] = (__bf16)0.0f; }
}

// ---- fused LSTM step -------------------------------------------------------
// g = [h, x_t] @ W_all^T + b ; gate math + c/h update wave-locally.
// Grid: 8 blocks x 256 threads = 64 waves; wave w owns hidden cols [16w,16w+16)
// across all four gate tiles (n, n+H, n+2H, n+3H), so i/f/o/cc meet in-lane.
__global__ __launch_bounds__(256) void lstm_step_kernel(
        int t,
        const __bf16* __restrict__ Wbf,   // [4H, 2H]
        const __bf16* __restrict__ xe,    // [S, B, H]
        const __bf16* __restrict__ hprev, // [B, H]
        __bf16* __restrict__ hnext,       // [B, H]
        float* __restrict__ cstate,       // [B, H]
        const float* __restrict__ bi, const float* __restrict__ bfv,
        const float* __restrict__ bo, const float* __restrict__ bc,
        __bf16* __restrict__ ys,          // [S, B, H]
        float* __restrict__ outH, float* __restrict__ outC) {
    const int lane = threadIdx.x & 31;
    const int wave = (blockIdx.x * blockDim.x + threadIdx.x) >> 5; // 0..63
    const int jw   = wave * 16;
    const int nl   = lane & 15;
    const int hi   = lane >> 4;
    const int m    = nl;                 // A row = batch
    const int kbase = hi * 16;           // A-frag K sub-offset (contiguous, matches B)
    const int ktop  = hi * 16;           // B-frag K sub-offset (contiguous, matches A)

    v8f acc[4];
    #pragma unroll
    for (int g = 0; g < 4; ++g)
        #pragma unroll
        for (int r = 0; r < 8; ++r) acc[g][r] = 0.0f;

    const __bf16* hrow = hprev + (size_t)m * H_;
    const __bf16* xrow = xe + ((size_t)t * B_ + m) * H_;

    for (int kk = 0; kk < 64; ++kk) {
        const int k0 = kk * 32;
        const __bf16* asrc = (k0 < H_) ? (hrow + k0) : (xrow + (k0 - H_));
        union { v16bf v; uint4 q[2]; } af;
        af.q[0] = *(const uint4*)(asrc + kbase);
        af.q[1] = *(const uint4*)(asrc + 8 + kbase);
        #pragma unroll
        for (int g = 0; g < 4; ++g) {
            const __bf16* wrow =
                Wbf + ((size_t)(g * H_ + jw + nl)) * (2 * H_) + k0 + ktop;
            union { v16bf v; uint4 q[2]; } bf_;
            bf_.q[0] = *(const uint4*)(wrow);
            bf_.q[1] = *(const uint4*)(wrow + 8);
            acc[g] = __builtin_amdgcn_wmma_f32_16x16x32_bf16(
                false, af.v, false, bf_.v, (short)0, acc[g], false, false);
        }
    }

    const int j = jw + nl;
    const float vbi = bi[j], vbf = bfv[j], vbo = bo[j], vbc = bc[j];
    #pragma unroll
    for (int r = 0; r < 8; ++r) {
        const int mb = r + 8 * hi;       // C-frag row = batch
        float si = sigmoidf_(acc[0][r] + vbi);
        float sf = sigmoidf_(acc[1][r] + vbf);
        float so = sigmoidf_(acc[2][r] + vbo);
        float tc = tanhf(acc[3][r] + vbc);
        float cOld = cstate[mb * H_ + j];
        float cNew = sf * cOld + si * tc;
        float hNew = so * tanhf(cNew);
        cstate[mb * H_ + j] = cNew;
        __bf16 hb = (__bf16)hNew;
        hnext[mb * H_ + j] = hb;
        ys[((size_t)t * B_ + mb) * H_ + j] = hb;
        if (t == S_ - 1) {
            outH[mb * H_ + j] = hNew;
            outC[mb * H_ + j] = cNew;
        }
    }
}

// ---- tied output projection ------------------------------------------------
// logits[b][s][v] = sum_h ys[s][b][h]*emb[v][h] + fc_b[v].
// Block: 256 thr (8 waves). Wave = 1 M-tile (one s, all 16 batches) x 8 N-tiles
// (128 vocab). B tile (128 x 32k bf16) staged in LDS, double buffered, shared
// by all 8 waves; filled with async global->LDS loads.
// Grid: (250 n-groups, 32 m-groups).
#define PROJ_NT 8
#define LDSPAD 40   // row stride in bf16: 80 B = 16B-aligned, bank-spread
__global__ __launch_bounds__(256) void proj_kernel(
        const __bf16* __restrict__ ysbf, const __bf16* __restrict__ embbf,
        const float* __restrict__ fcb, float* __restrict__ out) {
    __shared__ __bf16 ldsB[2][16 * PROJ_NT][LDSPAD];   // 2 x 10 KB

    const int tid  = threadIdx.x;
    const int lane = tid & 31;
    const int wv   = tid >> 5;                  // 0..7
    const int mtile = blockIdx.y * 8 + wv;      // 0..255 (== s)
    const int n0    = blockIdx.x * (16 * PROJ_NT);
    const int nl = lane & 15;
    const int hi = lane >> 4;
    const int kbase = hi * 16, ktop = hi * 16;
    const int m = nl;

    // Staging assignment: thread -> (row, 16-elem column chunk of 32-k slab).
    const int srow = tid >> 1;                  // 0..127
    const int scol = (tid & 1) * 16;            // 0 or 16
    const __bf16* gsrc = embbf + (size_t)(n0 + srow) * H_ + scol;

    v8f acc[PROJ_NT];
    #pragma unroll
    for (int tl = 0; tl < PROJ_NT; ++tl)
        #pragma unroll
        for (int r = 0; r < 8; ++r) acc[tl][r] = 0.0f;

    const __bf16* arow = ysbf + ((size_t)mtile * B_ + m) * H_;

    auto stage = [&](int kk, int buf) {
        const __bf16* g = gsrc + kk * 32;
        __bf16* l = &ldsB[buf][srow][scol];
        copy16_g2l(g, l);
        copy16_g2l(g + 8, l + 8);
    };

    stage(0, 0);
    ASYNC_WAIT();
    __syncthreads();

    for (int kk = 0; kk < 32; ++kk) {
        const int buf = kk & 1;
        if (kk + 1 < 32) stage(kk + 1, buf ^ 1);   // overlap fill with math

        const int k0 = kk * 32;
        __builtin_prefetch(arow + k0 + 32, 0, 1);  // global_prefetch next A
        union { v16bf v; uint4 q[2]; } af;
        af.q[0] = *(const uint4*)(arow + k0 + kbase);
        af.q[1] = *(const uint4*)(arow + k0 + 8 + kbase);
        #pragma unroll
        for (int tl = 0; tl < PROJ_NT; ++tl) {
            const __bf16* lrow = &ldsB[buf][tl * 16 + nl][ktop];
            union { v16bf v; uint4 q[2]; } bf_;
            bf_.q[0] = *(const uint4*)(lrow);      // ds_load_b128
            bf_.q[1] = *(const uint4*)(lrow + 8);
            acc[tl] = __builtin_amdgcn_wmma_f32_16x16x32_bf16(
                false, af.v, false, bf_.v, (short)0, acc[tl], false, false);
        }
        ASYNC_WAIT();          // next buffer landed
        __syncthreads();       // all waves done with current buffer
    }

    const int s = mtile;
    #pragma unroll
    for (int tl = 0; tl < PROJ_NT; ++tl) {
        const int v = n0 + tl * 16 + nl;
        const float bias = fcb[v];
        #pragma unroll
        for (int r = 0; r < 8; ++r) {
            const int b = r + 8 * hi;   // C-frag row = batch
            out[(size_t)b * S_ * V_ + (size_t)s * V_ + v] = acc[tl][r] + bias;
        }
    }
}

// ---------------------------------------------------------------------------

extern "C" void kernel_launch(void* const* d_in, const int* in_sizes, int n_in,
                              void* d_out, int out_size, void* d_ws, size_t ws_size,
                              hipStream_t stream) {
    (void)in_sizes; (void)n_in; (void)out_size; (void)ws_size;

    const int*   x    = (const int*)  d_in[0];
    const float* emb  = (const float*)d_in[1];
    const float* W_i  = (const float*)d_in[2];
    const float* b_i  = (const float*)d_in[3];
    const float* W_f  = (const float*)d_in[4];
    const float* b_f  = (const float*)d_in[5];
    const float* W_o  = (const float*)d_in[6];
    const float* b_o  = (const float*)d_in[7];
    const float* W_c  = (const float*)d_in[8];
    const float* b_c  = (const float*)d_in[9];
    const float* fc_b = (const float*)d_in[10];

    char* ws = (char*)d_ws;
    __bf16* embbf = (__bf16*)(ws + OFF_EMBBF);
    __bf16* Wbf   = (__bf16*)(ws + OFF_WBF);
    __bf16* xebf  = (__bf16*)(ws + OFF_XEBF);
    __bf16* ysbf  = (__bf16*)(ws + OFF_YSBF);
    __bf16* hbf   = (__bf16*)(ws + OFF_HBF);
    float*  cst   = (float*) (ws + OFF_CST);

    float* out  = (float*)d_out;
    float* outH = out + (size_t)B_ * S_ * V_;
    float* outC = outH + (size_t)B_ * H_;

    {
        int n4 = (V_ * H_) / 4;                       // 8,192,000
        cast_bf16_kernel<<<n4 / 256, 256, 0, stream>>>(emb, embbf, n4);
    }
    pack_w_kernel<<<4 * H_, 256, 0, stream>>>(W_i, W_f, W_o, W_c, Wbf);
    embed_kernel<<<S_ * B_, 256, 0, stream>>>(x, emb, xebf);
    init_state_kernel<<<(B_ * H_ + 255) / 256, 256, 0, stream>>>(cst, hbf);

    // Sequential LSTM scan; h double-buffered (bf16) across step kernels.
    for (int t = 0; t < S_; ++t) {
        const __bf16* hprev = hbf + (size_t)(t & 1) * B_ * H_;
        __bf16*       hnext = hbf + (size_t)((t + 1) & 1) * B_ * H_;
        lstm_step_kernel<<<8, 256, 0, stream>>>(
            t, Wbf, xebf, hprev, hnext, cst,
            b_i, b_f, b_o, b_c, ysbf, outH, outC);
    }

    // Tied output projection (268 GFLOP; LDS-staged B, 524 MB logits store).
    proj_kernel<<<dim3(250, 32), 256, 0, stream>>>(ysbf, embbf, fc_b, out);
}
